// FibonacciAttention_48361331752988
// MI455X (gfx1250) — matrix operand order repaired and checked by reading, not compiled, not run on hardware
//
#include <hip/hip_runtime.h>
#include <hip/hip_bf16.h>

typedef __attribute__((ext_vector_type(16))) _Float16 v16h;
typedef __attribute__((ext_vector_type(8)))  float    v8f;

#define DIM 768
#define NH  12
#define HD  64
#define BB  2
#define SS  2048
#define NTOK (BB*SS)

// Distinct fibonacci offsets (plus self) that fit in S=2048.
__constant__ int g_fib[17] = {0,1,2,3,5,8,13,21,34,55,89,144,233,377,610,987,1597};

// ---- CDNA5 async-to-LDS helpers ------------------------------------------
// ISA 10.2: LDS aperture truncates the flat address to addr[31:0], so the low
// 32 bits of a __shared__ pointer are the DS byte address.
__device__ __forceinline__ uint32_t lds_addr_of(const void* p) {
    return (uint32_t)(uintptr_t)p;
}
__device__ __forceinline__ void async_copy_b128(uint32_t lds, const void* g) {
    asm volatile("global_load_async_to_lds_b128 %0, %1, off"
                 :: "v"(lds), "v"((unsigned long long)(uintptr_t)g)
                 : "memory");
}
// Async loads retire in order (ISA 8 §4.1): waiting ASYNCcnt<=6 with 6 newer
// copies in flight proves the previous tile's 6 copies have landed.
__device__ __forceinline__ void wait_async_le6() {
    asm volatile("s_wait_asynccnt 0x6" ::: "memory");
}
__device__ __forceinline__ void wait_async_le0() {
    asm volatile("s_wait_asynccnt 0x0" ::: "memory");
}

// ---- one-shot f32 -> f16 conversion (hoists cvt out of GEMM hot loop) ----
__global__ __launch_bounds__(256)
void cvt_f32_f16(const float4* __restrict__ in, _Float16* __restrict__ out) {
    const int i = blockIdx.x * blockDim.x + threadIdx.x;   // n % 1024 == 0
    const float4 v = in[i];
    _Float16* o = out + (size_t)i * 4;
    o[0] = (_Float16)v.x; o[1] = (_Float16)v.y;
    o[2] = (_Float16)v.z; o[3] = (_Float16)v.w;
}

// ---- WMMA fragment loaders (ISA 7.12.2 wave32 layouts) -------------------
__device__ __forceinline__ v16h load_afrag(const _Float16 (&S)[128][72],
                                           int row, int kb0, int lane) {
    v16h a;
    const int m    = row + (lane & 15);
    const int koff = (lane >> 4) * 8;
    #pragma unroll
    for (int h = 0; h < 16; ++h) {
        const int kk = (h < 8) ? (koff + h) : (koff + h + 8);
        a[h] = S[m][kb0 + kk];
    }
    return a;
}
__device__ __forceinline__ v16h load_bfrag(const _Float16 (&S)[64][72],
                                           int row, int kb0, int lane) {
    v16h b;
    const int n  = row + (lane & 15);
    const int kb = (lane >> 4) * 8;
    #pragma unroll
    for (int h = 0; h < 16; ++h) b[h] = S[n][kb0 + ((h < 8) ? (kb + h) : (kb + h + 8))];
    return b;
}

// C[M,N] = A[M,K](f16) @ W[N,K](f16)^T + bias[N]
// Block tile 128x64, K-step 64, double-buffered async LDS pipeline.
// 8 waves, each wave owns a 32x32 tile -> 8 WMMAs per K-step.
// MODE 0: row-major f32 store.  MODE 1: scatter into q/k/v [B,H,S,HD] f32.
template<int MODE>
__global__ __launch_bounds__(256)
void gemm_wmma(const _Float16* __restrict__ A, const _Float16* __restrict__ W,
               const float* __restrict__ bias, float* __restrict__ C,
               float* __restrict__ qb, float* __restrict__ kb, float* __restrict__ vb,
               int M, int N, int K)
{
    __shared__ _Float16 As[2][128][72];   // double-buffered 128 x 64k (padded)
    __shared__ _Float16 Bs[2][64][72];

    const int tid  = threadIdx.x;
    const int lane = tid & 31;
    const int w    = tid >> 5;            // 8 waves
    const int wm   = w & 3;               // 4 wave rows (4*32 = 128)
    const int wn   = w >> 2;              // 2 wave cols (2*32 = 64)
    const int m0   = blockIdx.x * 128;
    const int n0   = blockIdx.y * 64;

    const uint32_t as_base = lds_addr_of(&As[0][0][0]);
    const uint32_t bs_base = lds_addr_of(&Bs[0][0][0]);
    const uint32_t as_sz   = (uint32_t)sizeof(As[0]);   // 18432 B
    const uint32_t bs_sz   = (uint32_t)sizeof(Bs[0]);   //  9216 B

    v8f acc[2][2] = {};

    // Stage one 128x64 A tile (1024 16B chunks, 4/thread) + 64x64 W tile
    // (512 chunks, 2/thread) into LDS buffer `buf` via async copies.
    auto issue_tile = [&](int kt, int buf) {
        const int k0 = kt * 64;
        const uint32_t ab = as_base + (uint32_t)buf * as_sz;
        const uint32_t bb = bs_base + (uint32_t)buf * bs_sz;
        #pragma unroll
        for (int i = 0; i < 4; ++i) {
            const int c = tid + 256 * i;
            const int r = c >> 3, col = (c & 7) * 8;
            async_copy_b128(ab + (uint32_t)(r * 72 + col) * 2,
                            A + (size_t)(m0 + r) * K + k0 + col);
        }
        #pragma unroll
        for (int i = 0; i < 2; ++i) {
            const int c = tid + 256 * i;
            const int r = c >> 3, col = (c & 7) * 8;
            async_copy_b128(bb + (uint32_t)(r * 72 + col) * 2,
                            W + (size_t)(n0 + r) * K + k0 + col);
        }
    };

    const int NK = K / 64;                // 12 for K=768
    issue_tile(0, 0);

    for (int kt = 0; kt < NK; ++kt) {
        const int buf = kt & 1;
        if (kt + 1 < NK) {
            issue_tile(kt + 1, buf ^ 1);  // overlap next tile with this compute
            wait_async_le6();             // previous 6 (this tile) have landed
        } else {
            wait_async_le0();
        }
        if (kt + 2 < NK)                  // L2 prefetch two stages ahead
            __builtin_prefetch(A + (size_t)(m0 + (tid >> 3)) * K + (kt + 2) * 64, 0, 1);
        __syncthreads();

        #pragma unroll
        for (int kh = 0; kh < 2; ++kh) {  // two 32-wide K slices
            const int kb0 = kh * 32;
            const v16h a0 = load_afrag(As[buf], wm * 32,      kb0, lane);
            const v16h a1 = load_afrag(As[buf], wm * 32 + 16, kb0, lane);
            const v16h b0 = load_bfrag(Bs[buf], wn * 32,      kb0, lane);
            const v16h b1 = load_bfrag(Bs[buf], wn * 32 + 16, kb0, lane);
            acc[0][0] = __builtin_amdgcn_wmma_f32_16x16x32_f16(
                            false, a0, false, b0, (short)0, acc[0][0], false, false);
            acc[0][1] = __builtin_amdgcn_wmma_f32_16x16x32_f16(
                            false, a0, false, b1, (short)0, acc[0][1], false, false);
            acc[1][0] = __builtin_amdgcn_wmma_f32_16x16x32_f16(
                            false, a1, false, b0, (short)0, acc[1][0], false, false);
            acc[1][1] = __builtin_amdgcn_wmma_f32_16x16x32_f16(
                            false, a1, false, b1, (short)0, acc[1][1], false, false);
        }
        __syncthreads();                  // before buf is re-targeted at kt+2
    }

    // C/D layout: n = lane&15, m = r + 8*(lane>>4)
    const int nl = lane & 15;
    const int mh = (lane >> 4) * 8;
    #pragma unroll
    for (int si = 0; si < 2; ++si) {
        #pragma unroll
        for (int sj = 0; sj < 2; ++sj) {
            const int gn = n0 + wn * 32 + sj * 16 + nl;
            const float bv = bias[gn];
            #pragma unroll
            for (int r = 0; r < 8; ++r) {
                const int gm = m0 + wm * 32 + si * 16 + mh + r;
                const float val = acc[si][sj][r] + bv;
                if (MODE == 0) {
                    C[(size_t)gm * N + gn] = val;
                } else {
                    const int which = gn / DIM;          // 0=q 1=k 2=v
                    const int col   = gn - which * DIM;
                    const int h2    = col >> 6;
                    const int d     = col & 63;
                    const int b     = gm >> 11;
                    const int s     = gm & (SS - 1);
                    float* dst = (which == 0) ? qb : (which == 1) ? kb : vb;
                    dst[(((size_t)(b * NH + h2) * SS) + s) * HD + d] = val;
                }
            }
        }
    }
}

// Sparse fibonacci attention: one wave32 per (b,h,s).
// Lanes 0..16 each own one fib offset; softmax across lanes via shuffles.
// Writes f16 context in [b, s, h*64+d] order (feeds output-proj GEMM).
__global__ __launch_bounds__(256)
void fib_attn(const float* __restrict__ qb, const float* __restrict__ kb,
              const float* __restrict__ vb, _Float16* __restrict__ ctx)
{
    const int lane = threadIdx.x & 31;
    const int g    = blockIdx.x * 8 + (threadIdx.x >> 5);
    const int s    = g & (SS - 1);
    const int bh   = g >> 11;                            // b*NH + h
    const size_t head_base = (size_t)bh * SS * HD;
    const float* q = qb + head_base + (size_t)s * HD;

    const int  off   = (lane < 17) ? g_fib[lane] : (SS + 1);
    const bool valid = (s - off) >= 0;

    float score = -__builtin_inff();
    if (valid) {
        const float4* q4 = (const float4*)q;
        const float4* k4 = (const float4*)(kb + head_base + (size_t)(s - off) * HD);
        float d = 0.f;
        #pragma unroll
        for (int t = 0; t < 16; ++t) {
            float4 a = q4[t], b = k4[t];
            d += a.x*b.x + a.y*b.y + a.z*b.z + a.w*b.w;
        }
        score = d * 0.125f;   // hd^-0.5
    }

    float mx = score;
    #pragma unroll
    for (int m = 16; m >= 1; m >>= 1) mx = fmaxf(mx, __shfl_xor(mx, m, 32));
    float p = valid ? __expf(score - mx) : 0.f;
    float denom = p;
    #pragma unroll
    for (int m = 16; m >= 1; m >>= 1) denom += __shfl_xor(denom, m, 32);
    p /= denom;   // diag offset 0 always valid

    float2 out = {0.f, 0.f};
    #pragma unroll
    for (int j = 0; j < 17; ++j) {
        const int oj = g_fib[j];
        if (s >= oj) {                                   // wave-uniform branch
            const float pj = __shfl(p, j, 32);
            const float2* v2 = (const float2*)(vb + head_base + (size_t)(s - oj) * HD);
            const float2 vv = v2[lane];
            out.x += pj * vv.x;
            out.y += pj * vv.y;
        }
    }

    const int h = bh % NH;
    const int b = bh / NH;
    _Float16* dst = ctx + (size_t)(b * SS + s) * DIM + h * HD + lane * 2;
    dst[0] = (_Float16)out.x;
    dst[1] = (_Float16)out.y;
}

extern "C" void kernel_launch(void* const* d_in, const int* in_sizes, int n_in,
                              void* d_out, int out_size, void* d_ws, size_t ws_size,
                              hipStream_t stream) {
    const float* x     = (const float*)d_in[0];   // [B,S,DIM]
    const float* w_qkv = (const float*)d_in[1];   // [3*DIM, DIM]
    const float* b_qkv = (const float*)d_in[2];   // [3*DIM]
    const float* w_out = (const float*)d_in[3];   // [DIM, DIM]
    const float* b_out = (const float*)d_in[4];   // [DIM]
    float* out = (float*)d_out;

    // workspace layout
    float* ws = (float*)d_ws;
    const size_t HBUF = (size_t)NTOK * DIM;       // 3,145,728
    float* qbuf = ws;
    float* kbuf = ws + HBUF;
    float* vbuf = ws + 2 * HBUF;
    _Float16* xh   = (_Float16*)(ws + 3 * HBUF);              // NTOK*DIM halves
    _Float16* wqh  = xh  + HBUF;                              // 3*DIM*DIM
    _Float16* woh  = wqh + (size_t)3 * DIM * DIM;             // DIM*DIM
    _Float16* ctxh = woh + (size_t)DIM * DIM;                 // NTOK*DIM

    dim3 blk(256);

    // 0) one-shot f32 -> f16 packs (element counts all % 1024 == 0)
    cvt_f32_f16<<<dim3((unsigned)(HBUF / 1024)), blk, 0, stream>>>((const float4*)x, xh);
    cvt_f32_f16<<<dim3((unsigned)((size_t)3 * DIM * DIM / 1024)), blk, 0, stream>>>((const float4*)w_qkv, wqh);
    cvt_f32_f16<<<dim3((unsigned)((size_t)DIM * DIM / 1024)), blk, 0, stream>>>((const float4*)w_out, woh);

    // 1) QKV projection: [4096,768] @ [2304,768]^T -> q/k/v [B,H,S,hd] f32
    dim3 g1(NTOK / 128, (3 * DIM) / 64);
    gemm_wmma<1><<<g1, blk, 0, stream>>>(xh, wqh, b_qkv, nullptr,
                                         qbuf, kbuf, vbuf, NTOK, 3 * DIM, DIM);

    // 2) Sparse fibonacci attention (one wave per (b,h,s)) -> ctx f16
    dim3 g2((BB * NH * SS) / 8);
    fib_attn<<<g2, blk, 0, stream>>>(qbuf, kbuf, vbuf, ctxh);

    // 3) Output projection: ctx [4096,768] @ [768,768]^T + b -> d_out f32
    dim3 g3(NTOK / 128, DIM / 64);
    gemm_wmma<0><<<g3, blk, 0, stream>>>(ctxh, woh, b_out, out,
                                         nullptr, nullptr, nullptr, NTOK, DIM, DIM);
}
